// GATTest_68083821576593
// MI455X (gfx1250) — matrix operand order repaired and checked by reading, not compiled, not run on hardware
//
#include <hip/hip_runtime.h>
#include <hip/hip_fp16.h>

// ---------------------------------------------------------------- constants
constexpr int NN   = 8192;           // nodes
constexpr int EE   = 32768;          // edges (before self loops)
constexpr int MM   = 2048;           // masked nodes
constexpr int DD   = 768;            // input dim
constexpr int HH   = 8;              // heads layer 1
constexpr int CC   = 768;            // per-head channels
constexpr int FF   = HH * CC;        // 6144
constexpr int EPAD = EE + NN;        // edges + self loops = 40960
constexpr float NEG_SLOPE = 0.2f;

typedef __attribute__((ext_vector_type(16))) _Float16 v16h;
typedef __attribute__((ext_vector_type(8)))  float    v8f;

union Frag16 { v16h h; uint32_t u[8]; };
struct alignas(16) H8 { _Float16 h[8]; };   // 16B chunk of f16
struct alignas(8)  H4 { _Float16 h[4]; };   // 8B  chunk of f16

// ---------------------------------------------------------------- WMMA GEMM
// C[M,N] = A[M,K]*B[K,N] (+bias[N]); A,B f16 in memory, f32 accumulate.
// M%128==0, N%128==0, K%32==0 (and K >= 2*BK). LDS ping-pong double
// buffering; peeled so the steady-state body is branch-free:
//   fill(p^1, kt) ; compute(p) ; barrier ; swap     ... ; compute(last)
template <typename OutT>
__global__ __launch_bounds__(256)
void wmma_gemm_f16(const _Float16* __restrict__ A, const _Float16* __restrict__ B,
                   OutT* __restrict__ C, const float* __restrict__ bias,
                   int Mdim, int Ndim, int Kdim)
{
    constexpr int BM = 128, BN = 128, BK = 32, BKP = 40;   // BKP: +16B row pad
    __shared__ _Float16 As[2][BM][BKP];   // [buf][row][k]
    __shared__ _Float16 Bs[2][BN][BKP];   // [buf][col][k] (transposed on fill)

    const int tid  = threadIdx.x;
    const int lane = tid & 31;
    const int wave = tid >> 5;
    const int wr   = wave & 1;         // 2 wave-rows (64 rows each)
    const int wc   = wave >> 1;        // 4 wave-cols (32 cols each)
    const int bm   = blockIdx.y * BM;
    const int bn   = blockIdx.x * BN;

    // per-thread fill geometry
    const int ar0 = tid >> 2;                 // A chunk 0: row, k8 = (tid&3)*8
    const int ak0 = (tid & 3) << 3;
    const int ar1 = (tid + 256) >> 2;         // A chunk 1
    const int ak1 = ((tid + 256) & 3) << 3;
    const int bkq = (tid >> 5) << 2;          // B 4x4 block: k quad
    const int bn4 = (tid & 31) << 2;          //              n quad

    v8f acc[4][2];
#pragma unroll
    for (int i = 0; i < 4; ++i)
#pragma unroll
        for (int j = 0; j < 2; ++j)
#pragma unroll
            for (int e = 0; e < 8; ++e) acc[i][j][e] = 0.0f;

    const int kbaseA = (lane >> 4) << 3;   // 0 or 8  (A fragment K interleave)
    const int khB    = (lane >> 4) << 3;   // 0 or 8  (B fragment K interleave)

    // global -> registers -> LDS buffer p (no state survives this block)
    auto fill = [&](int p, int kt) {
        H8 a0 = *(const H8*)&A[(size_t)(bm + ar0) * Kdim + kt + ak0];
        H8 a1 = *(const H8*)&A[(size_t)(bm + ar1) * Kdim + kt + ak1];
        H4 b0 = *(const H4*)&B[(size_t)(kt + bkq + 0) * Ndim + bn + bn4];
        H4 b1 = *(const H4*)&B[(size_t)(kt + bkq + 1) * Ndim + bn + bn4];
        H4 b2 = *(const H4*)&B[(size_t)(kt + bkq + 2) * Ndim + bn + bn4];
        H4 b3 = *(const H4*)&B[(size_t)(kt + bkq + 3) * Ndim + bn + bn4];
        *(H8*)&As[p][ar0][ak0] = a0;
        *(H8*)&As[p][ar1][ak1] = a1;
#pragma unroll
        for (int i = 0; i < 4; ++i) {          // in-register 4x4 transpose
            H4 t;
            t.h[0] = b0.h[i]; t.h[1] = b1.h[i];
            t.h[2] = b2.h[i]; t.h[3] = b3.h[i];
            *(H4*)&Bs[p][bn4 + i][bkq] = t;
        }
    };

    // fragments from buffer p + 8 WMMAs; A fragments loaded per-mt to keep
    // register pressure low (B fragments reused across all 4 mt steps)
    auto compute = [&](int p) {
        Frag16 bfr[2];
#pragma unroll
        for (int nt = 0; nt < 2; ++nt) {
            const uint32_t* brow =
                (const uint32_t*)&Bs[p][wc * 32 + nt * 16 + (lane & 15)][0];
            *(uint4*)&bfr[nt].u[0] = *(const uint4*)(brow + (khB >> 1));
            *(uint4*)&bfr[nt].u[4] = *(const uint4*)(brow + 8 + (khB >> 1));
        }
#pragma unroll
        for (int mt = 0; mt < 4; ++mt) {
            Frag16 afr;
            const uint32_t* arow =
                (const uint32_t*)&As[p][wr * 64 + mt * 16 + (lane & 15)][0];
            *(uint4*)&afr.u[0] = *(const uint4*)(arow + (kbaseA >> 1));
            *(uint4*)&afr.u[4] = *(const uint4*)(arow + 8 + (kbaseA >> 1));
            acc[mt][0] = __builtin_amdgcn_wmma_f32_16x16x32_f16(
                false, afr.h, false, bfr[0].h, (short)0, acc[mt][0], false, false);
            acc[mt][1] = __builtin_amdgcn_wmma_f32_16x16x32_f16(
                false, afr.h, false, bfr[1].h, (short)0, acc[mt][1], false, false);
        }
    };

    fill(0, 0);
    __syncthreads();

    int p = 0;
    for (int kt = BK; kt < Kdim; kt += BK) {   // steady state: branch-free body
        fill(p ^ 1, kt);
        compute(p);
        __syncthreads();
        p ^= 1;
    }
    compute(p);                                // last tile (no fill)

    // ---- epilogue: lane 0-15: M=v, N=lane; lane 16-31: M=8+v
#pragma unroll
    for (int mt = 0; mt < 4; ++mt) {
        int row = bm + wr * 64 + mt * 16 + ((lane >> 4) << 3);
#pragma unroll
        for (int nt = 0; nt < 2; ++nt) {
            int col = bn + wc * 32 + nt * 16 + (lane & 15);
            float bv = bias ? bias[col] : 0.0f;
#pragma unroll
            for (int v = 0; v < 8; ++v)
                C[(size_t)(row + v) * Ndim + col] = (OutT)(acc[mt][nt][v] + bv);
        }
    }
}

// ---------------------------------------------------------------- helpers
__device__ __forceinline__ uint32_t enc_f32(float f) {
    uint32_t u = __float_as_uint(f);
    return (u & 0x80000000u) ? ~u : (u | 0x80000000u);
}
__device__ __forceinline__ float dec_f32(uint32_t u) {
    uint32_t b = (u & 0x80000000u) ? (u ^ 0x80000000u) : ~u;
    return __uint_as_float(b);
}
__device__ __forceinline__ void edge_nodes(const int* __restrict__ ei, int e,
                                           int& s, int& d) {
    if (e < EE) { s = ei[e]; d = ei[EE + e]; }   // edge_index[0][e], [1][e]
    else        { s = d = e - EE; }              // self loops
}

__global__ void zero_f32(float* __restrict__ p, size_t n) {
    size_t i = ((size_t)blockIdx.x * blockDim.x + threadIdx.x) * 4;
    if (i < n) *(float4*)(p + i) = make_float4(0.f, 0.f, 0.f, 0.f);
}

// fp32 -> fp16 cast, 4 elems/thread (b128 in, b64 out)
__global__ void cast_f32_f16(const float* __restrict__ in,
                             _Float16* __restrict__ out, size_t n) {
    size_t i = ((size_t)blockIdx.x * blockDim.x + threadIdx.x) * 4;
    if (i >= n) return;
    float4 v = *(const float4*)(in + i);
    H4 p;
    p.h[0] = (_Float16)v.x; p.h[1] = (_Float16)v.y;
    p.h[2] = (_Float16)v.z; p.h[3] = (_Float16)v.w;
    *(H4*)(out + i) = p;
}

// ---------------------------------------------------------------- attention
// a_src[n,h] = sum_c h[n,h,c]*att_src[h,c]; one wave per (n,h); h in f16
__global__ void att_dot_kernel(const _Float16* __restrict__ hbuf,
                               const float* __restrict__ att_src,
                               const float* __restrict__ att_dst,
                               float* __restrict__ asrc, float* __restrict__ adst,
                               int n_nodes, int heads, int ch)
{
    int wid  = (int)((blockIdx.x * (size_t)blockDim.x + threadIdx.x) >> 5);
    int lane = threadIdx.x & 31;
    if (wid >= n_nodes * heads) return;
    int n = wid / heads, h = wid % heads;
    const _Float16* hp = hbuf + (size_t)n * heads * ch + (size_t)h * ch;
    float s0 = 0.f, s1 = 0.f;
    for (int c = lane; c < ch; c += 32) {
        float v = (float)hp[c];
        s0 += v * att_src[h * ch + c];
        s1 += v * att_dst[h * ch + c];
    }
    for (int off = 16; off > 0; off >>= 1) {
        s0 += __shfl_down(s0, off, 32);
        s1 += __shfl_down(s1, off, 32);
    }
    if (lane == 0) { asrc[wid] = s0; adst[wid] = s1; }
}

// e_raw = leaky_relu(a_src[src]+a_dst[dst]); segment max via encoded uint
__global__ void edge_pass1(const int* __restrict__ ei,
                           const float* __restrict__ asrc,
                           const float* __restrict__ adst,
                           float* __restrict__ eraw, unsigned* __restrict__ menc,
                           int heads)
{
    int idx = blockIdx.x * blockDim.x + threadIdx.x;
    if (idx >= EPAD * heads) return;
    int e = idx / heads, h = idx % heads;
    int s, d; edge_nodes(ei, e, s, d);
    float v = asrc[s * heads + h] + adst[d * heads + h];
    v = v > 0.f ? v : NEG_SLOPE * v;
    eraw[idx] = v;
    atomicMax(&menc[d * heads + h], enc_f32(v));
}

// e_exp = exp(e_raw - m[dst]); segment sum into denom
__global__ void edge_pass2(const int* __restrict__ ei,
                           const float* __restrict__ eraw,
                           float* __restrict__ eexp,
                           const unsigned* __restrict__ menc,
                           float* __restrict__ denom, int heads)
{
    int idx = blockIdx.x * blockDim.x + threadIdx.x;
    if (idx >= EPAD * heads) return;
    int e = idx / heads, h = idx % heads;
    int s, d; edge_nodes(ei, e, s, d);
    (void)s;
    float ex = __expf(eraw[idx] - dec_f32(menc[d * heads + h]));
    eexp[idx] = ex;
    atomicAdd(&denom[d * heads + h], ex);
}

// out[dst] += h[src]*alpha ; one block per edge; h in f16, accumulate f32
__global__ void edge_aggregate(const int* __restrict__ ei,
                               const _Float16* __restrict__ hbuf,
                               const float* __restrict__ eexp,
                               const float* __restrict__ denom,
                               float* __restrict__ out, int heads, int ch)
{
    int e = blockIdx.x;
    int s, d; edge_nodes(ei, e, s, d);
    int F = heads * ch;
    const _Float16* hs = hbuf + (size_t)s * F;
    float*          od = out  + (size_t)d * F;
    for (int i = threadIdx.x; i < F; i += blockDim.x) {
        int h = i / ch;
        float alpha = eexp[e * heads + h] / (denom[d * heads + h] + 1e-16f);
        atomicAdd(&od[i], (float)hs[i] * alpha);
    }
}

// layer-1 finalize: f32 agg + bias -> ELU -> f16 (GEMM2 input)
__global__ void bias_elu_to_f16(const float* __restrict__ in,
                                const float* __restrict__ bias,
                                _Float16* __restrict__ out, size_t total, int F)
{
    size_t i = ((size_t)blockIdx.x * blockDim.x + threadIdx.x) * 4;
    if (i >= total) return;
    float4 v = *(const float4*)(in + i);
    int col = (int)(i % F);
    float a = v.x + bias[col], b = v.y + bias[col + 1];
    float c = v.z + bias[col + 2], d = v.w + bias[col + 3];
    a = a > 0.f ? a : (__expf(a) - 1.f);
    b = b > 0.f ? b : (__expf(b) - 1.f);
    c = c > 0.f ? c : (__expf(c) - 1.f);
    d = d > 0.f ? d : (__expf(d) - 1.f);
    H4 p; p.h[0] = (_Float16)a; p.h[1] = (_Float16)b;
    p.h[2] = (_Float16)c; p.h[3] = (_Float16)d;
    *(H4*)(out + i) = p;
}

// layer-2 finalize: f32 agg += bias (in place, stays f32 for gather)
__global__ void bias_add(float* __restrict__ buf, const float* __restrict__ bias,
                         size_t total, int F)
{
    size_t idx = (size_t)blockIdx.x * blockDim.x + threadIdx.x;
    if (idx >= total) return;
    buf[idx] += bias[idx % F];
}

// xcat[m] = concat(f16(agg2[mask[m]]), xh[mask[m]])  (f16 output for fc GEMM)
__global__ void gather_concat(const float* __restrict__ agg2,
                              const _Float16* __restrict__ xh,
                              const int* __restrict__ mask_idx,
                              _Float16* __restrict__ xcat)
{
    int m = blockIdx.x;
    int node = mask_idx[m];
    const float*    a = agg2 + (size_t)node * CC;
    const _Float16* b = xh   + (size_t)node * DD;
    _Float16*       o = xcat + (size_t)m * (2 * CC);
    for (int i = threadIdx.x; i < CC; i += blockDim.x) o[i]      = (_Float16)a[i];
    for (int i = threadIdx.x; i < DD; i += blockDim.x) o[CC + i] = b[i];
}

// out[m,0:2] = fc_out[m] @ cls_w + cls_b ; one wave per row
__global__ void cls_kernel(const float* __restrict__ fc_out,
                           const float* __restrict__ cls_w,
                           const float* __restrict__ cls_b,
                           float* __restrict__ out)
{
    int wid  = (int)((blockIdx.x * (size_t)blockDim.x + threadIdx.x) >> 5);
    int lane = threadIdx.x & 31;
    if (wid >= MM) return;
    const float* r = fc_out + (size_t)wid * CC;
    float s0 = 0.f, s1 = 0.f;
    for (int c = lane; c < CC; c += 32) {
        float v = r[c];
        s0 += v * cls_w[c * 2 + 0];
        s1 += v * cls_w[c * 2 + 1];
    }
    for (int off = 16; off > 0; off >>= 1) {
        s0 += __shfl_down(s0, off, 32);
        s1 += __shfl_down(s1, off, 32);
    }
    if (lane == 0) { out[wid * 2 + 0] = s0 + cls_b[0]; out[wid * 2 + 1] = s1 + cls_b[1]; }
}

// ---------------------------------------------------------------- launch
extern "C" void kernel_launch(void* const* d_in, const int* in_sizes, int n_in,
                              void* d_out, int out_size, void* d_ws, size_t ws_size,
                              hipStream_t stream)
{
    const float* x        = (const float*)d_in[0];
    const int*   ei       = (const int*)  d_in[1];
    const int*   mask_idx = (const int*)  d_in[2];
    const float* W1       = (const float*)d_in[3];
    const float* att_src1 = (const float*)d_in[4];
    const float* att_dst1 = (const float*)d_in[5];
    const float* b1       = (const float*)d_in[6];
    const float* W2       = (const float*)d_in[7];
    const float* att_src2 = (const float*)d_in[8];
    const float* att_dst2 = (const float*)d_in[9];
    const float* b2       = (const float*)d_in[10];
    const float* fc_w     = (const float*)d_in[11];
    const float* fc_b     = (const float*)d_in[12];
    const float* cls_w    = (const float*)d_in[13];
    const float* cls_b    = (const float*)d_in[14];
    float* out = (float*)d_out;

    // -------- workspace carve-up (bytes, 256B aligned)
    char* wsb = (char*)d_ws;
    size_t off = 0;
    auto alloc = [&](size_t bytes) {
        void* p = wsb + off;
        off += (bytes + 255) & ~(size_t)255;
        return p;
    };
    _Float16* h1h    = (_Float16*)alloc((size_t)NN * FF * 2);   // x@W1 (f16)
    float*    agg1   = (float*)   alloc((size_t)NN * FF * 4);   // layer-1 atomic acc
    _Float16* xh     = (_Float16*)alloc((size_t)NN * DD * 2);
    _Float16* W1h    = (_Float16*)alloc((size_t)DD * FF * 2);
    _Float16* W2h    = (_Float16*)alloc((size_t)FF * CC * 2);
    _Float16* fcwh   = (_Float16*)alloc((size_t)2 * CC * CC * 2);
    float*    asrc1  = (float*)alloc((size_t)NN * HH * 4);
    float*    adst1  = (float*)alloc((size_t)NN * HH * 4);
    float*    eraw1  = (float*)alloc((size_t)EPAD * HH * 4);
    float*    eexp1  = (float*)alloc((size_t)EPAD * HH * 4);
    float*    denom1 = (float*)alloc((size_t)NN * HH * 4);
    float*    menc1f = (float*)alloc((size_t)NN * HH * 4);
    float*    asrc2  = (float*)alloc((size_t)NN * 4);
    float*    adst2  = (float*)alloc((size_t)NN * 4);
    float*    eraw2  = (float*)alloc((size_t)EPAD * 4);
    float*    eexp2  = (float*)alloc((size_t)EPAD * 4);
    float*    denom2 = (float*)alloc((size_t)NN * 4);
    float*    menc2f = (float*)alloc((size_t)NN * 4);
    _Float16* xcath  = (_Float16*)alloc((size_t)MM * 2 * CC * 2);
    float*    fc_out = (float*)alloc((size_t)MM * CC * 4);
    // region reuse: h1h dead after layer-1 aggregation -> ELU output (f16)
    _Float16* g1h  = h1h;
    // agg1 dead after bias_elu_to_f16 -> layer-2 h and aggregation buffers
    _Float16* h2h  = (_Float16*)agg1;
    float*    agg2 = (float*)((char*)agg1 + (((size_t)NN * CC * 2 + 255) & ~(size_t)255));
    unsigned* menc1 = (unsigned*)menc1f;
    unsigned* menc2 = (unsigned*)menc2f;

    auto g4 = [](size_t n) { return (unsigned)((n / 4 + 255) / 256); };

    // -------- one-time f32 -> f16 operand casts
    cast_f32_f16<<<g4((size_t)NN * DD), 256, 0, stream>>>(x,    xh,   (size_t)NN * DD);
    cast_f32_f16<<<g4((size_t)DD * FF), 256, 0, stream>>>(W1,   W1h,  (size_t)DD * FF);
    cast_f32_f16<<<g4((size_t)FF * CC), 256, 0, stream>>>(W2,   W2h,  (size_t)FF * CC);
    cast_f32_f16<<<g4((size_t)2 * CC * CC), 256, 0, stream>>>(fc_w, fcwh, (size_t)2 * CC * CC);

    // ============ layer 1: GAT(768 -> 8 x 768) ============
    zero_f32<<<g4((size_t)NN * FF), 256, 0, stream>>>(agg1,   (size_t)NN * FF);
    zero_f32<<<g4((size_t)NN * HH), 256, 0, stream>>>(denom1, (size_t)NN * HH);
    zero_f32<<<g4((size_t)NN * HH), 256, 0, stream>>>(menc1f, (size_t)NN * HH);

    wmma_gemm_f16<_Float16><<<dim3(FF / 128, NN / 128), 256, 0, stream>>>(
        xh, W1h, h1h, nullptr, NN, FF, DD);

    att_dot_kernel<<<(NN * HH * 32 + 255) / 256, 256, 0, stream>>>(
        h1h, att_src1, att_dst1, asrc1, adst1, NN, HH, CC);
    edge_pass1<<<(EPAD * HH + 255) / 256, 256, 0, stream>>>(
        ei, asrc1, adst1, eraw1, menc1, HH);
    edge_pass2<<<(EPAD * HH + 255) / 256, 256, 0, stream>>>(
        ei, eraw1, eexp1, menc1, denom1, HH);
    edge_aggregate<<<EPAD, 256, 0, stream>>>(ei, h1h, eexp1, denom1, agg1, HH, CC);
    bias_elu_to_f16<<<g4((size_t)NN * FF), 256, 0, stream>>>(
        agg1, b1, g1h, (size_t)NN * FF, FF);

    // ============ layer 2: GAT(6144 -> 768), 1 head ============
    wmma_gemm_f16<_Float16><<<dim3(CC / 128, NN / 128), 256, 0, stream>>>(
        g1h, W2h, h2h, nullptr, NN, CC, FF);
    zero_f32<<<g4((size_t)NN * CC), 256, 0, stream>>>(agg2,   (size_t)NN * CC);
    zero_f32<<<g4((size_t)NN), 256, 0, stream>>>(denom2, (size_t)NN);
    zero_f32<<<g4((size_t)NN), 256, 0, stream>>>(menc2f, (size_t)NN);

    att_dot_kernel<<<(NN * 32 + 255) / 256, 256, 0, stream>>>(
        h2h, att_src2, att_dst2, asrc2, adst2, NN, 1, CC);
    edge_pass1<<<(EPAD + 255) / 256, 256, 0, stream>>>(
        ei, asrc2, adst2, eraw2, menc2, 1);
    edge_pass2<<<(EPAD + 255) / 256, 256, 0, stream>>>(
        ei, eraw2, eexp2, menc2, denom2, 1);
    edge_aggregate<<<EPAD, 256, 0, stream>>>(ei, h2h, eexp2, denom2, agg2, 1, CC);
    bias_add<<<(unsigned)(((size_t)NN * CC + 255) / 256), 256, 0, stream>>>(
        agg2, b2, (size_t)NN * CC, CC);

    // ============ head: gather -> fc -> classifier ============
    gather_concat<<<MM, 256, 0, stream>>>(agg2, xh, mask_idx, xcath);
    wmma_gemm_f16<float><<<dim3(CC / 128, MM / 128), 256, 0, stream>>>(
        xcath, fcwh, fc_out, fc_b, MM, CC, 2 * CC);
    cls_kernel<<<(MM * 32 + 255) / 256, 256, 0, stream>>>(
        fc_out, cls_w, cls_b, out);

    (void)in_sizes; (void)n_in; (void)out_size; (void)ws_size;
}
